// DynamicAttention_10316511445031
// MI455X (gfx1250) — matrix operand order repaired and checked by reading, not compiled, not run on hardware
//
#include <hip/hip_runtime.h>

typedef __attribute__((ext_vector_type(16))) _Float16 v16h;
typedef __attribute__((ext_vector_type(8)))  float    v8f;

// ---------------- problem constants ----------------
#define BATCH   8
#define HW      96
#define CH      192
#define HEADS   8
#define HD      24
#define WS      12
#define WS2     144          // tokens per window
#define NWIN    512          // 8 * 8 * 8 windows
#define XS      200          // LDS row stride (halfs) for x/Q/K (192 + pad)
#define PST     168          // LDS row stride (halfs) for P (K padded to 160)
#define VTS     168          // LDS row stride (halfs) for V^T (tokens padded to 160)
#define SST     148          // LDS row stride (floats) for logits
#define OFS     196          // LDS row stride (floats) for proj output staging

// LDS carve (bytes) for attn kernel
#define OFF_X    0           // Xh [144][200] f16, reused as P [144][168] f16
#define OFF_Q    57600       // Qh [144][200] f16 (pre-scaled by 24^-0.5)
#define OFF_K    115200      // Kh [144][200] f16
#define OFF_VT   172800      // Vt [200][168] f16 (row=dim, col=token; tokens 144..159 zero)
#define OFF_S    240000      // Sf [144][148] f32, reused as Sh [144][32] f16 staging
#define OFF_INV  325248      // RowInv [144] f32
#define LDS_ATTN (OFF_INV + WS2*4)             // 325824 B
#define LDS_PROJ (WS2*XS*2 + WS2*OFS*4)        // 57600 + 112896 = 170496 B

// ---------------- WMMA helpers (wave32, 16x16x32 f16) ----------------
__device__ __forceinline__ v8f wmma_f16(v16h a, v16h b, v8f c) {
    return __builtin_amdgcn_wmma_f32_16x16x32_f16(false, a, false, b, (short)0, c, false, false);
}

// A (16x32 f16) from row-major [M][ld] LDS, full K
__device__ __forceinline__ v16h frag_a(const _Float16* M, int ld, int m0, int k0, int lane) {
    const _Float16* p = M + (m0 + (lane & 15)) * ld + k0 + ((lane < 16) ? 0 : 16);
    v16h a;
#pragma unroll
    for (int i = 0; i < 8; ++i) a[i] = p[i];
#pragma unroll
    for (int i = 0; i < 8; ++i) a[8 + i] = p[8 + i];
    return a;
}

// A fragment for K-dim 24 (zero-padded to 32): only upper-half lanes zero their 2nd group
__device__ __forceinline__ v16h frag_a_hd24(const _Float16* M, int ld, int m0, int lane) {
    bool lo = lane < 16;
    const _Float16* p = M + (m0 + (lane & 15)) * ld + (lo ? 0 : 16);
    v16h a;
#pragma unroll
    for (int i = 0; i < 8; ++i) a[i] = p[i];              // K 0..7 / 16..23 : valid
    if (lo) {
#pragma unroll
        for (int i = 0; i < 8; ++i) a[8 + i] = p[8 + i]; // K 8..15 : valid
    } else {
#pragma unroll
        for (int i = 0; i < 8; ++i) a[8 + i] = (_Float16)0.f; // K 24..31 : pad
    }
    return a;
}

// B = W^T, W row-major [N][ldw]: 16 contiguous K per lane
__device__ __forceinline__ v16h frag_bT(const _Float16* W, int ldw, int k0, int n0, int lane) {
    const _Float16* p = W + (size_t)(n0 + (lane & 15)) * ldw + k0 + ((lane < 16) ? 0 : 16);
    v16h b;
#pragma unroll
    for (int i = 0; i < 16; ++i) b[i] = p[i];
    return b;
}

// B^T fragment for K-dim 24 (padded to 32): upper-half lanes valid only for first 8
__device__ __forceinline__ v16h frag_bT_hd24(const _Float16* M, int ld, int n0, int lane) {
    bool lo = lane < 16;
    const _Float16* p = M + (n0 + (lane & 15)) * ld + (lo ? 0 : 16);
    v16h b;
    if (lo) {
#pragma unroll
        for (int i = 0; i < 16; ++i) b[i] = p[i];         // K 0..15
    } else {
#pragma unroll
        for (int i = 0; i < 8; ++i)  b[i] = p[i];         // K 16..23
#pragma unroll
        for (int i = 8; i < 16; ++i) b[i] = (_Float16)0.f; // K 24..31 pad
    }
    return b;
}

// ---------------- kernel 0: weight fp32 -> fp16 ----------------
__global__ void __launch_bounds__(256) k_convert(const float* __restrict__ qkvw,
                                                 const float* __restrict__ projw,
                                                 _Float16* __restrict__ qkvw_h,
                                                 _Float16* __restrict__ projw_h) {
    int i = blockIdx.x * 256 + threadIdx.x;
    if (i < 3 * CH * CH) qkvw_h[i] = (_Float16)qkvw[i];
    if (i < CH * CH)     projw_h[i] = (_Float16)projw[i];
}

// ---------------- kernel 1: relative-position bias matrix [8][144][144] ----------------
__global__ void __launch_bounds__(256) k_bias(const float* __restrict__ btab,
                                              float* __restrict__ biasm) {
    int i = blockIdx.x * 256 + threadIdx.x;
    if (i >= HEADS * WS2 * WS2) return;
    int h = i / (WS2 * WS2), r = i % (WS2 * WS2);
    int q = r / WS2, k = r % WS2;
    int idx = (q / WS - k / WS + WS - 1) * (2 * WS - 1) + (q % WS - k % WS + WS - 1);
    biasm[i] = btab[idx * HEADS + h];
}

// ---------------- kernel 2: depthwise 3x3 LPE branch -> d_out ----------------
__global__ void __launch_bounds__(256) k_lpe(const float* __restrict__ x,
                                             const float* __restrict__ noise,
                                             const float* __restrict__ lw,
                                             const float* __restrict__ lb,
                                             const float* __restrict__ lscale,
                                             float* __restrict__ out) {
    size_t i = (size_t)blockIdx.x * 256 + threadIdx.x;
    if (i >= (size_t)BATCH * HW * HW * CH) return;
    int c  = (int)(i % CH);
    int t  = (int)(i / CH);
    int wq = t % HW; int t2 = t / HW;
    int hq = t2 % HW; int b = t2 / HW;
    float acc = lb[c];
#pragma unroll
    for (int ky = 0; ky < 3; ++ky) {
        int hy = hq + ky - 1;
        if (hy < 0 || hy >= HW) continue;
#pragma unroll
        for (int kx = 0; kx < 3; ++kx) {
            int wx = wq + kx - 1;
            if (wx < 0 || wx >= HW) continue;
            acc += x[((size_t)b * HW * HW + hy * HW + wx) * CH + c] * lw[c * 9 + ky * 3 + kx];
        }
    }
    out[i] = acc * lscale[0] * noise[(size_t)b * HW * HW + hq * HW + wq];
}

// ---------------- kernel 3: per-window fused QKV + attention ----------------
__global__ void __launch_bounds__(256) k_attn(const float* __restrict__ x,
                                              const _Float16* __restrict__ qkvw_h,
                                              const float* __restrict__ qkvb,
                                              const float* __restrict__ biasm,
                                              _Float16* __restrict__ attn_out) {
    extern __shared__ char smem[];
    _Float16* Xh = (_Float16*)(smem + OFF_X);
    _Float16* Qh = (_Float16*)(smem + OFF_Q);
    _Float16* Kh = (_Float16*)(smem + OFF_K);
    _Float16* Vt = (_Float16*)(smem + OFF_VT);
    float*    Sf = (float*)   (smem + OFF_S);
    float*    RowInv = (float*)(smem + OFF_INV);
    _Float16* Ph = Xh;                       // [144][PST]
    _Float16* Sh = (_Float16*)Sf;            // [144][32] f16 output staging

    int w  = blockIdx.x;
    int b  = w >> 6, wy = (w >> 3) & 7, wx = w & 7;
    int r0 = wy * WS, c0 = wx * WS;
    int tid = threadIdx.x, lane = tid & 31, wid = tid >> 5;
    const float scale = 0.20412414523193154f;   // 24^-0.5 (folded into Q)

    // ---- stage x window fp32->fp16 (float4, coalesced) + zero Vt token pad ----
    const float* xb = x + (size_t)b * HW * HW * CH;
    for (int i = tid; i < WS2 * (CH / 4); i += 256) {
        int t = i / (CH / 4), c4 = (i % (CH / 4)) * 4;
        int r = r0 + t / WS, cl = c0 + t % WS;
        float4 v = *(const float4*)(xb + ((size_t)(r * HW + cl)) * CH + c4);
        _Float16* d = Xh + t * XS + c4;
        d[0] = (_Float16)v.x; d[1] = (_Float16)v.y; d[2] = (_Float16)v.z; d[3] = (_Float16)v.w;
    }
    for (int i = tid; i < 200 * 16; i += 256) {       // Vt tokens 144..159 = 0
        int r = i >> 4, c = WS2 + (i & 15);
        Vt[r * VTS + c] = (_Float16)0.f;
    }
    __syncthreads();

    // ---- QKV GEMM: [144,192] x [192,576] (9x36 tiles, 6 K-chunks) ----
    for (int tile = wid; tile < 9 * 36; tile += 8) {
        int mt = tile / 36, nt = tile % 36;
        int m0 = mt * 16, n0 = nt * 16;
        float bias = qkvb[n0 + (lane & 15)];
        v8f acc;
#pragma unroll
        for (int i = 0; i < 8; ++i) acc[i] = bias;
#pragma unroll
        for (int kc = 0; kc < 6; ++kc) {
            v16h a  = frag_a(Xh, XS, m0, kc * 32, lane);
            v16h bb = frag_bT(qkvw_h, CH, kc * 32, n0, lane);
            acc = wmma_f16(a, bb, acc);
        }
        int n = lane & 15, moff = (lane < 16) ? 0 : 8;
        int seg = n0 / CH, cc = n0 % CH + n;
        if (seg == 0) {          // Q, pre-scaled
#pragma unroll
            for (int i = 0; i < 8; ++i) Qh[(m0 + moff + i) * XS + cc] = (_Float16)(acc[i] * scale);
        } else if (seg == 1) {   // K
#pragma unroll
            for (int i = 0; i < 8; ++i) Kh[(m0 + moff + i) * XS + cc] = (_Float16)acc[i];
        } else {                 // V transposed: row=dim, col=token (8 contiguous halfs)
#pragma unroll
            for (int i = 0; i < 8; ++i) Vt[cc * VTS + (m0 + moff + i)] = (_Float16)acc[i];
        }
    }
    __syncthreads();

    for (int h = 0; h < HEADS; ++h) {
        int hc = h * HD;
        const float* bh = biasm + h * (WS2 * WS2);

        // ---- S = Q K^T + bias (81 tiles, K=24 padded to 32) ----
        for (int tile = wid; tile < 81; tile += 8) {
            int mt = tile / 9, nt = tile % 9;
            int m0 = mt * 16, n0 = nt * 16;
            v8f acc;
#pragma unroll
            for (int i = 0; i < 8; ++i) acc[i] = 0.f;
            v16h a  = frag_a_hd24(Qh + hc, XS, m0, lane);
            v16h bb = frag_bT_hd24(Kh + hc, XS, n0, lane);
            acc = wmma_f16(a, bb, acc);
            int n = lane & 15, moff = (lane < 16) ? 0 : 8;
            const float* bp = bh + (m0 + moff) * WS2 + n0 + n;
            float* sp = Sf + (m0 + moff) * SST + n0 + n;
#pragma unroll
            for (int i = 0; i < 8; ++i) sp[i * SST] = acc[i] + bp[i * WS2];
        }
        __syncthreads();

        // ---- softmax, 2-pass, deferred normalization ----
        if (tid < WS2) {
            float* srow = Sf + tid * SST;
            float mx = -1e30f;
            for (int k = 0; k < WS2; ++k) mx = fmaxf(mx, srow[k]);
            float sum = 0.f;
            _Float16* prow = Ph + tid * PST;
            for (int k = 0; k < WS2; ++k) {
                float e = __expf(srow[k] - mx);
                sum += e;
                prow[k] = (_Float16)e;
            }
            for (int k = WS2; k < 160; ++k) prow[k] = (_Float16)0.f;
            RowInv[tid] = 1.f / sum;
        }
        __syncthreads();

        // ---- out = (P V) * RowInv -> Sh staging (9x2 tiles, K=160) ----
        for (int tile = wid; tile < 18; tile += 8) {
            int mt = tile >> 1, nt = tile & 1;
            int m0 = mt * 16;
            v8f acc;
#pragma unroll
            for (int i = 0; i < 8; ++i) acc[i] = 0.f;
#pragma unroll
            for (int kc = 0; kc < 5; ++kc) {
                v16h a  = frag_a(Ph, PST, m0, kc * 32, lane);
                v16h bb = frag_bT(Vt, VTS, kc * 32, hc + nt * 16, lane);
                acc = wmma_f16(a, bb, acc);
            }
            int n = lane & 15, moff = (lane < 16) ? 0 : 8;
            int col = nt * 16 + n;
#pragma unroll
            for (int i = 0; i < 8; ++i) {
                int m = m0 + moff + i;
                Sh[m * 32 + col] = (_Float16)(acc[i] * RowInv[m]);
            }
        }
        __syncthreads();

        // ---- coalesced copy: Sh[144][32] -> attn_out layout [w][h][144][24] ----
        _Float16* dsth = attn_out + ((size_t)w * HEADS + h) * WS2 * HD;
        for (int i = tid; i < WS2 * (HD / 2); i += 256) {
            int t = i / (HD / 2), c2 = (i % (HD / 2)) * 2;
            *(unsigned int*)(dsth + t * HD + c2) = *(const unsigned int*)(Sh + t * 32 + c2);
        }
        __syncthreads();
    }
}

// ---------------- kernel 4: output projection + window reverse + add LPE ----------------
__global__ void __launch_bounds__(256) k_proj(const _Float16* __restrict__ attn,
                                              const _Float16* __restrict__ projw_h,
                                              const float* __restrict__ projb,
                                              float* __restrict__ out) {
    extern __shared__ char smem[];
    _Float16* Ah = (_Float16*)smem;                 // [144][XS]
    float*    Of = (float*)(smem + WS2 * XS * 2);   // [144][OFS]

    int w  = blockIdx.x;
    int b  = w >> 6, wy = (w >> 3) & 7, wx = w & 7;
    int tid = threadIdx.x, lane = tid & 31, wid = tid >> 5;

    // gather per-head-major scratch into [token][C] f16 LDS (coalesced global reads)
    const _Float16* src = attn + (size_t)w * HEADS * WS2 * HD;
    for (int i = tid; i < HEADS * WS2 * (HD / 2); i += 256) {
        int h = i / (WS2 * (HD / 2)), r = i % (WS2 * (HD / 2));
        int t = r / (HD / 2), c2 = (r % (HD / 2)) * 2;
        *(unsigned int*)(Ah + t * XS + h * HD + c2) =
            *(const unsigned int*)(src + ((size_t)h * WS2 + t) * HD + c2);
    }
    __syncthreads();

    // GEMM [144,192] x [192,192] + bias -> Of (9x12 tiles, 6 K-chunks)
    for (int tile = wid; tile < 9 * 12; tile += 8) {
        int mt = tile / 12, nt = tile % 12;
        int m0 = mt * 16, n0 = nt * 16;
        float pb = projb[n0 + (lane & 15)];
        v8f acc;
#pragma unroll
        for (int i = 0; i < 8; ++i) acc[i] = pb;
#pragma unroll
        for (int kc = 0; kc < 6; ++kc) {
            v16h a  = frag_a(Ah, XS, m0, kc * 32, lane);
            v16h bb = frag_bT(projw_h, CH, kc * 32, n0, lane);
            acc = wmma_f16(a, bb, acc);
        }
        int n = lane & 15, moff = (lane < 16) ? 0 : 8;
        float* p = Of + (m0 + moff) * OFS + n0 + n;
#pragma unroll
        for (int i = 0; i < 8; ++i) p[i * OFS] = acc[i];
    }
    __syncthreads();

    // coalesced float4 RMW epilogue: window reverse + add LPE already in d_out
    for (int i = tid; i < WS2 * (CH / 4); i += 256) {
        int t = i / (CH / 4), c4 = (i % (CH / 4)) * 4;
        int r  = wy * WS + t / WS;
        int cl = wx * WS + t % WS;
        float* g = out + ((size_t)b * HW * HW + r * HW + cl) * CH + c4;
        const float* l = Of + t * OFS + c4;
        float4 v = *(float4*)g;
        v.x += l[0]; v.y += l[1]; v.z += l[2]; v.w += l[3];
        *(float4*)g = v;
    }
}

// ---------------- host launch ----------------
extern "C" void kernel_launch(void* const* d_in, const int* in_sizes, int n_in,
                              void* d_out, int out_size, void* d_ws, size_t ws_size,
                              hipStream_t stream) {
    (void)in_sizes; (void)n_in; (void)out_size; (void)ws_size;
    const float* x_seq = (const float*)d_in[0];
    const float* noise = (const float*)d_in[1];
    const float* qkvw  = (const float*)d_in[2];
    const float* qkvb  = (const float*)d_in[3];
    const float* projw = (const float*)d_in[4];
    const float* projb = (const float*)d_in[5];
    const float* lpew  = (const float*)d_in[6];
    const float* lpeb  = (const float*)d_in[7];
    const float* lpes  = (const float*)d_in[8];
    const float* btab  = (const float*)d_in[9];
    float* out = (float*)d_out;

    char* ws = (char*)d_ws;
    _Float16* qkvw_h  = (_Float16*)ws;                    // 221184 B
    _Float16* projw_h = (_Float16*)(ws + 221184);         //  73728 B
    float*    biasm   = (float*)   (ws + 294912);         // 663552 B
    _Float16* attn    = (_Float16*)(ws + 958464);         // 512*8*144*24*2 = 28.3 MB

    k_convert<<<(3 * CH * CH + 255) / 256, 256, 0, stream>>>(qkvw, projw, qkvw_h, projw_h);
    k_bias<<<(HEADS * WS2 * WS2 + 255) / 256, 256, 0, stream>>>(btab, biasm);

    size_t total = (size_t)BATCH * HW * HW * CH;
    k_lpe<<<(int)((total + 255) / 256), 256, 0, stream>>>(x_seq, noise, lpew, lpeb, lpes, out);

    k_attn<<<NWIN, 256, LDS_ATTN, stream>>>(x_seq, qkvw_h, qkvb, biasm, attn);

    k_proj<<<NWIN, 256, LDS_PROJ, stream>>>(attn, projw_h, projb, out);
}
